// GraphTransformerLayer_12146167513684
// MI455X (gfx1250) — matrix operand order repaired and checked by reading, not compiled, not run on hardware
//
#include <hip/hip_runtime.h>
#include <hip/hip_bf16.h>

#define N_NODES 2048
#define DMODEL  1024
#define NHEAD   16
#define DKH     64
#define DFF     4096
#define NET     4

typedef __bf16 bf16_t;
typedef bf16_t v16bf __attribute__((ext_vector_type(16)));
typedef float  v8f   __attribute__((ext_vector_type(8)));
typedef unsigned short u16;
typedef unsigned int   u32;

#define MODE_BF16      0
#define MODE_BF16_GELU 1
#define MODE_F32       2

// ---- gfx1250 async global->LDS copy (ASYNCcnt) with graceful fallback ----
#if defined(__has_builtin)
#if __has_builtin(__builtin_amdgcn_global_load_async_to_lds_b128)
#define HAVE_ASYNC_LDS 1
#endif
#endif
#ifndef HAVE_ASYNC_LDS
#define HAVE_ASYNC_LDS 0
#endif

#if HAVE_ASYNC_LDS
// builtin signature: (v4i addrspace(1)* src, v4i addrspace(3)* dst, imm offset, imm cpol)
typedef int v4i __attribute__((vector_size(16)));
#define ASYNC_CP16(gsrc, ldst)                                                  \
  __builtin_amdgcn_global_load_async_to_lds_b128(                               \
      (__attribute__((address_space(1))) v4i*)(void*)(gsrc),                    \
      (__attribute__((address_space(3))) v4i*)(void*)(ldst), 0, 0)

template <int N>
static __device__ __forceinline__ void wait_asyncN() {
#if __has_builtin(__builtin_amdgcn_s_wait_asynccnt)
  __builtin_amdgcn_s_wait_asynccnt(N);
#else
  asm volatile("s_wait_asynccnt %0" ::"i"(N) : "memory");
#endif
}
#endif

// ---------------------------------------------------------------- helpers

static __device__ __forceinline__ u16 f2bf(float x) {
  union { float f; u32 u; } c; c.f = x;
  u32 r = c.u + 0x7FFFu + ((c.u >> 16) & 1u);   // round-to-nearest-even
  return (u16)(r >> 16);
}

static __device__ __forceinline__ v16bf make_frag(uint4 a, uint4 b) {
  union { v16bf v; uint4 q[2]; } u;
  u.q[0] = a; u.q[1] = b;
  return u.v;
}

// A-matrix fragment (16x32 bf16) from row-major [rows][ld] storage.
// ISA: lanes 0-15 (M=lane) hold K=0..7 / K=16..23; lanes 16-31 hold K=8..15 / 24..31.
static __device__ __forceinline__ v16bf fragA(const u16* s, int ld, int row0, int k0) {
  const int lane = threadIdx.x & 31;
  const int m = lane & 15, hi = lane >> 4;
  const u16* p = s + (size_t)(row0 + m) * ld + k0 + (hi ? 8 : 0);
  uint4 a = *(const uint4*)p;
  uint4 b = *(const uint4*)(p + 16);
  return make_frag(a, b);
}

// B-matrix fragment (32x16 bf16) from "n-major" [n][k] storage with stride ld.
// ISA: lane n (0-15) holds K=0..7 / K=16..23; lanes 16-31 hold K=8..15 / 24..31.
static __device__ __forceinline__ v16bf fragB(const u16* s, int ld, int n0, int k0) {
  const int lane = threadIdx.x & 31;
  const int n = lane & 15, hi = lane >> 4;
  const u16* p = s + (size_t)(n0 + n) * ld + k0 + (hi ? 8 : 0);
  uint4 a = *(const uint4*)p;
  uint4 b = *(const uint4*)(p + 16);
  return make_frag(a, b);
}

static __device__ __forceinline__ v8f wmma_bf16(v16bf a, v16bf b, v8f c) {
  return __builtin_amdgcn_wmma_f32_16x16x32_bf16(false, a, false, b, (short)0, c, false, false);
}

// ---------------------------------------------------------------- f32 -> bf16

__global__ __launch_bounds__(256)
void to_bf16_kernel(const float* __restrict__ in, u16* __restrict__ out, int n) {
  int i = (blockIdx.x * 256 + threadIdx.x) * 4;
  if (i >= n) return;
  float4 f = *(const float4*)(in + i);
  u32 lo = (u32)f2bf(f.x) | ((u32)f2bf(f.y) << 16);
  u32 hi = (u32)f2bf(f.z) | ((u32)f2bf(f.w) << 16);
  uint2 r; r.x = lo; r.y = hi;
  *(uint2*)(out + i) = r;
}

// f32 [Kk][Nn] -> bf16 transposed [Nn][Kk]; tiled 64x64 via LDS.
__global__ __launch_bounds__(256)
void transpose_bf16_kernel(const float* __restrict__ in, u16* __restrict__ out,
                           int Kk, int Nn) {
  __shared__ u16 t[64][72];
  const int tid = threadIdx.x;
  const int kb = blockIdx.y * 64, nb = blockIdx.x * 64;
  const int r = tid >> 2;            // 0..63
  const int cs = (tid & 3) * 16;     // 0,16,32,48

  const float* src = in + (size_t)(kb + r) * Nn + nb + cs;
  #pragma unroll
  for (int e = 0; e < 16; e += 4) {
    float4 f = *(const float4*)(src + e);
    t[cs + e + 0][r] = f2bf(f.x);
    t[cs + e + 1][r] = f2bf(f.y);
    t[cs + e + 2][r] = f2bf(f.z);
    t[cs + e + 3][r] = f2bf(f.w);
  }
  __syncthreads();
  u16* dst = out + (size_t)(nb + r) * Kk + kb + cs;
  #pragma unroll
  for (int e = 0; e < 16; e += 8)
    *(uint4*)(dst + e) = *(const uint4*)&t[r][cs + e];
}

// ---------------------------------------------------------------- tiled WMMA GEMM
// C[M][Nn] = scale * (A[M][Kk] @ B[Kk][Nn]) (+ bias); B supplied TRANSPOSED
// as BT[Nn][Kk]. Block: 256 threads (8 waves), tile 128(M) x 128(N), K-step 32.
// Wave tile 32x64 -> 8 WMMAs per K-step. Async copies double-buffered so the
// next tile streams into LDS behind the current tile's WMMAs.

__global__ __launch_bounds__(256)
void gemm_bf16_kernel(const u16* __restrict__ A, const u16* __restrict__ BT,
                      const float* __restrict__ bias, void* __restrict__ out,
                      int M, int Nn, int Kk, float scale, int mode) {
  __shared__ u16 sA[2][128 * 40];    // [m][k], padded stride 40
  __shared__ u16 sBT[2][128 * 40];   // [n][k], padded stride 40

  const int tid  = threadIdx.x;
  const int lane = tid & 31;
  const int wid  = tid >> 5;
  const int n0 = blockIdx.x * 128;
  const int m0 = blockIdx.y * 128;
  const int mw = (wid >> 1) * 32;   // wave sub-tile: 32 rows
  const int nw = (wid & 1) * 64;    // 64 cols

  v8f acc[2][4] = {};

  const int ar = tid >> 1, as = (tid & 1) * 16;   // staging: 2 thr/row, 16 el each
  const u16* gA = A  + (size_t)(m0 + ar) * Kk + as;
  const u16* gB = BT + (size_t)(n0 + ar) * Kk + as;
  const int so = ar * 40 + as;

#if HAVE_ASYNC_LDS
  // prologue: stage tile 0 into buffer 0
  ASYNC_CP16(gA,     &sA[0][so]);
  ASYNC_CP16(gA + 8, &sA[0][so + 8]);
  ASYNC_CP16(gB,     &sBT[0][so]);
  ASYNC_CP16(gB + 8, &sBT[0][so + 8]);
  int buf = 0;
  for (int kb = 0; kb < Kk; kb += 32) {
    if (kb + 32 < Kk) {
      const u16* na = gA + kb + 32;
      const u16* nb = gB + kb + 32;
      __builtin_prefetch(na + 256, 0, 0);   // stream-ahead into GL2 (192MB L2)
      __builtin_prefetch(nb + 256, 0, 0);
      ASYNC_CP16(na,     &sA[buf ^ 1][so]);
      ASYNC_CP16(na + 8, &sA[buf ^ 1][so + 8]);
      ASYNC_CP16(nb,     &sBT[buf ^ 1][so]);
      ASYNC_CP16(nb + 8, &sBT[buf ^ 1][so + 8]);
      wait_asyncN<4>();   // 4 newest = next tile still in flight; current tile landed
    } else {
      wait_asyncN<0>();
    }
    __syncthreads();      // current buffer published by all waves

    v16bf a0 = fragA(sA[buf], 40, mw, 0);
    v16bf a1 = fragA(sA[buf], 40, mw + 16, 0);
    #pragma unroll
    for (int nt = 0; nt < 4; ++nt) {
      v16bf b = fragB(sBT[buf], 40, nw + nt * 16, 0);
      acc[0][nt] = wmma_bf16(a0, b, acc[0][nt]);
      acc[1][nt] = wmma_bf16(a1, b, acc[1][nt]);
    }
    __syncthreads();      // protect this buffer before it is overwritten
    buf ^= 1;
  }
#else
  for (int kb = 0; kb < Kk; kb += 32) {
    const u16* ga = gA + kb;
    const u16* gb = gB + kb;
    __builtin_prefetch(ga + 256, 0, 0);
    __builtin_prefetch(gb + 256, 0, 0);
    uint4 qa0 = *(const uint4*)ga;
    uint4 qa1 = *(const uint4*)(ga + 8);
    uint4 qb0 = *(const uint4*)gb;
    uint4 qb1 = *(const uint4*)(gb + 8);
    __syncthreads();
    *(uint4*)&sA[0][so]      = qa0;
    *(uint4*)&sA[0][so + 8]  = qa1;
    *(uint4*)&sBT[0][so]     = qb0;
    *(uint4*)&sBT[0][so + 8] = qb1;
    __syncthreads();

    v16bf a0 = fragA(sA[0], 40, mw, 0);
    v16bf a1 = fragA(sA[0], 40, mw + 16, 0);
    #pragma unroll
    for (int nt = 0; nt < 4; ++nt) {
      v16bf b = fragB(sBT[0], 40, nw + nt * 16, 0);
      acc[0][nt] = wmma_bf16(a0, b, acc[0][nt]);
      acc[1][nt] = wmma_bf16(a1, b, acc[1][nt]);
    }
  }
  __syncthreads();
#endif

  const int cn = lane & 15, hi = lane >> 4;
  #pragma unroll
  for (int mt = 0; mt < 2; ++mt) {
    #pragma unroll
    for (int nt = 0; nt < 4; ++nt) {
      #pragma unroll
      for (int r = 0; r < 8; ++r) {
        int row = m0 + mw + mt * 16 + r + 8 * hi;
        int col = n0 + nw + nt * 16 + cn;
        float v = acc[mt][nt][r] * scale;
        if (bias) v += bias[col];
        if (mode == MODE_BF16_GELU) v = 0.5f * v * (1.0f + erff(v * 0.70710678f));
        if (mode == MODE_F32) ((float*)out)[(size_t)row * Nn + col] = v;
        else                  ((u16*)out)[(size_t)row * Nn + col]   = f2bf(v);
      }
    }
  }
}

// ---------------------------------------------------------------- attention
// Flash-style, one wave per (head, 16-row block); 8 waves/WG share K/V tiles.
// Q pre-scaled by 1/sqrt(DKH); edge bias gathered from an LDS copy of the table.

__global__ __launch_bounds__(256)
void attn_kernel(const u16* __restrict__ Qb, const u16* __restrict__ Kb,
                 const u16* __restrict__ Vb, const int* __restrict__ eidx,
                 const float* __restrict__ ebias, u16* __restrict__ Ob) {
  __shared__ u16 sK[32 * 72];        // K tile, row-major [j][dk], pad 72
  __shared__ u16 sVT[64 * 40];       // V tile transposed [dk][j], pad 40
  __shared__ u16 sP[8][16 * 40];     // per-wave P scratch, row-major [i][j], pad 40
  __shared__ float sB[(NET + 1) * NHEAD];

  const int tid  = threadIdx.x;
  const int lane = tid & 31;
  const int wid  = tid >> 5;
  const int hd = blockIdx.x & 15;
  const int i0 = (blockIdx.x >> 4) * 128 + wid * 16;
  const int cn = lane & 15, hi = lane >> 4;

  if (tid < (NET + 1) * NHEAD) sB[tid] = ebias[tid];

  v16bf qa0 = fragA(Qb, DMODEL, i0, hd * DKH);
  v16bf qa1 = fragA(Qb, DMODEL, i0, hd * DKH + 32);

  float mrow[8], lrow[8];
  #pragma unroll
  for (int r = 0; r < 8; ++r) { mrow[r] = -3.4e38f; lrow[r] = 0.f; }
  v8f o[4] = {};

  const int sr = tid >> 3, ss = (tid & 7) * 8;  // staging: 8 thr/row, 8 el each
  u16* sPw = &sP[wid][0];

  for (int j0 = 0; j0 < N_NODES; j0 += 32) {
    const u16* gk = Kb + (size_t)(j0 + sr) * DMODEL + hd * DKH + ss;
    uint4 vq = *(const uint4*)(Vb + (size_t)(j0 + sr) * DMODEL + hd * DKH + ss);
#if HAVE_ASYNC_LDS
    __syncthreads();                       // previous iteration done with LDS
    ASYNC_CP16(gk, &sK[sr * 72 + ss]);
    u16 vt[8]; *(uint4*)vt = vq;
    #pragma unroll
    for (int e = 0; e < 8; ++e) sVT[(ss + e) * 40 + sr] = vt[e];
    wait_asyncN<0>();
    __syncthreads();
#else
    uint4 kq = *(const uint4*)gk;
    __syncthreads();
    *(uint4*)&sK[sr * 72 + ss] = kq;
    u16 vt[8]; *(uint4*)vt = vq;
    #pragma unroll
    for (int e = 0; e < 8; ++e) sVT[(ss + e) * 40 + sr] = vt[e];
    __syncthreads();
#endif

    // S = Q K^T (+ edge bias); K tile row-major == B-fragment layout directly
    v8f st[2];
    #pragma unroll
    for (int t = 0; t < 2; ++t) {
      v8f s = {};
      s = wmma_bf16(qa0, fragB(sK, 72, t * 16, 0), s);
      s = wmma_bf16(qa1, fragB(sK, 72, t * 16, 32), s);
      int j = j0 + t * 16 + cn;
      #pragma unroll
      for (int r = 0; r < 8; ++r) {
        int i = i0 + r + 8 * hi;
        int e = eidx[(size_t)i * N_NODES + j];
        s[r] += sB[e * NHEAD + hd];
      }
      st[t] = s;
    }

    // online softmax (rows live across 16-lane halves)
    float sc[8];
    #pragma unroll
    for (int r = 0; r < 8; ++r) {
      float mx = fmaxf(st[0][r], st[1][r]);
      #pragma unroll
      for (int d = 8; d >= 1; d >>= 1) mx = fmaxf(mx, __shfl_xor(mx, d, 32));
      float mnew = fmaxf(mrow[r], mx);
      sc[r] = __expf(mrow[r] - mnew);
      float p0 = __expf(st[0][r] - mnew);
      float p1 = __expf(st[1][r] - mnew);
      float rs = p0 + p1;
      #pragma unroll
      for (int d = 8; d >= 1; d >>= 1) rs += __shfl_xor(rs, d, 32);
      lrow[r] = lrow[r] * sc[r] + rs;
      mrow[r] = mnew;
      sPw[(r + 8 * hi) * 40 + cn]      = f2bf(p0);
      sPw[(r + 8 * hi) * 40 + 16 + cn] = f2bf(p1);
    }
    #pragma unroll
    for (int dt = 0; dt < 4; ++dt)
      #pragma unroll
      for (int r = 0; r < 8; ++r) o[dt][r] *= sc[r];

    // wave-private LDS round-trip: C-layout -> A-layout
    asm volatile("s_wait_dscnt 0x0" ::: "memory");
    v16bf pa = fragA(sPw, 40, 0, 0);
    #pragma unroll
    for (int dt = 0; dt < 4; ++dt)
      o[dt] = wmma_bf16(pa, fragB(sVT, 40, dt * 16, 0), o[dt]);
  }

  #pragma unroll
  for (int dt = 0; dt < 4; ++dt) {
    #pragma unroll
    for (int r = 0; r < 8; ++r) {
      float v = o[dt][r] / lrow[r];
      Ob[(size_t)(i0 + r + 8 * hi) * DMODEL + hd * DKH + dt * 16 + cn] = f2bf(v);
    }
  }
}

// ---------------------------------------------------------------- residual + LayerNorm

__global__ __launch_bounds__(256)
void add_ln_kernel(const float* __restrict__ x, const float* __restrict__ y,
                   const float* __restrict__ g, const float* __restrict__ b,
                   float* __restrict__ outf, u16* __restrict__ outb) {
  __shared__ float red[256];
  const int row = blockIdx.x;
  const int tid = threadIdx.x;
  const float* xr = x + (size_t)row * DMODEL;
  const float* yr = y + (size_t)row * DMODEL;

  float v[4]; float s = 0.f;
  #pragma unroll
  for (int e = 0; e < 4; ++e) { v[e] = xr[tid + e * 256] + yr[tid + e * 256]; s += v[e]; }
  red[tid] = s; __syncthreads();
  for (int d = 128; d > 0; d >>= 1) { if (tid < d) red[tid] += red[tid + d]; __syncthreads(); }
  float mu = red[0] * (1.f / DMODEL);
  __syncthreads();

  float q = 0.f;
  #pragma unroll
  for (int e = 0; e < 4; ++e) { float t = v[e] - mu; q += t * t; }
  red[tid] = q; __syncthreads();
  for (int d = 128; d > 0; d >>= 1) { if (tid < d) red[tid] += red[tid + d]; __syncthreads(); }
  float rstd = rsqrtf(red[0] * (1.f / DMODEL) + 1e-5f);

  #pragma unroll
  for (int e = 0; e < 4; ++e) {
    int c = tid + e * 256;
    float h = (v[e] - mu) * rstd * g[c] + b[c];
    outf[(size_t)row * DMODEL + c] = h;
    if (outb) outb[(size_t)row * DMODEL + c] = f2bf(h);
  }
}

// ---------------------------------------------------------------- launch

extern "C" void kernel_launch(void* const* d_in, const int* in_sizes, int n_in,
                              void* d_out, int out_size, void* d_ws, size_t ws_size,
                              hipStream_t stream) {
  const float* x     = (const float*)d_in[0];
  const int*   eidx  = (const int*)d_in[1];
  const float* Wq    = (const float*)d_in[2];
  const float* Wk    = (const float*)d_in[3];
  const float* Wv    = (const float*)d_in[4];
  const float* Wo    = (const float*)d_in[5];
  const float* bo    = (const float*)d_in[6];
  const float* ebias = (const float*)d_in[7];
  const float* w1    = (const float*)d_in[8];
  const float* b1    = (const float*)d_in[9];
  const float* w2    = (const float*)d_in[10];
  const float* b2    = (const float*)d_in[11];
  const float* g1    = (const float*)d_in[12];
  const float* be1   = (const float*)d_in[13];
  const float* g2    = (const float*)d_in[14];
  const float* be2   = (const float*)d_in[15];

  char* wsp = (char*)d_ws;
  size_t off = 0;
  auto alloc = [&](size_t bytes) -> char* {
    char* p = wsp + off; off += (bytes + 255) & ~(size_t)255; return p;
  };
  u16* xb   = (u16*)alloc((size_t)N_NODES * DMODEL * 2);
  u16* WqT  = (u16*)alloc((size_t)DMODEL * DMODEL * 2);   // [out][in]
  u16* WkT  = (u16*)alloc((size_t)DMODEL * DMODEL * 2);
  u16* WvT  = (u16*)alloc((size_t)DMODEL * DMODEL * 2);
  u16* WoT  = (u16*)alloc((size_t)DMODEL * DMODEL * 2);
  u16* w1T  = (u16*)alloc((size_t)DMODEL * DFF * 2);      // [DFF][DMODEL]
  u16* w2T  = (u16*)alloc((size_t)DFF * DMODEL * 2);      // [DMODEL][DFF]
  u16* Qb   = (u16*)alloc((size_t)N_NODES * DMODEL * 2);
  u16* Kb   = (u16*)alloc((size_t)N_NODES * DMODEL * 2);
  u16* Vb   = (u16*)alloc((size_t)N_NODES * DMODEL * 2);
  u16* Ob   = (u16*)alloc((size_t)N_NODES * DMODEL * 2);
  float* proj = (float*)alloc((size_t)N_NODES * DMODEL * 4);
  float* hf   = (float*)alloc((size_t)N_NODES * DMODEL * 4);
  u16*   hb   = (u16*)alloc((size_t)N_NODES * DMODEL * 2);
  u16*   ff1  = (u16*)alloc((size_t)N_NODES * DFF * 2);
  float* ff2  = (float*)alloc((size_t)N_NODES * DMODEL * 4);

  // activations: straight convert; weights: convert + transpose ([in][out]->[out][in])
  to_bf16_kernel<<<(N_NODES * DMODEL) / 1024, 256, 0, stream>>>(x, xb, N_NODES * DMODEL);
  auto cvtT = [&](const float* src, u16* dst, int Kk, int Nn) {
    transpose_bf16_kernel<<<dim3(Nn / 64, Kk / 64), 256, 0, stream>>>(src, dst, Kk, Nn);
  };
  cvtT(Wq, WqT, DMODEL, DMODEL);
  cvtT(Wk, WkT, DMODEL, DMODEL);
  cvtT(Wv, WvT, DMODEL, DMODEL);
  cvtT(Wo, WoT, DMODEL, DMODEL);
  cvtT(w1, w1T, DMODEL, DFF);
  cvtT(w2, w2T, DFF, DMODEL);

  // QKV projections (Q pre-scaled by 1/sqrt(DKH))
  gemm_bf16_kernel<<<dim3(DMODEL / 128, N_NODES / 128), 256, 0, stream>>>(
      xb, WqT, (const float*)nullptr, Qb, N_NODES, DMODEL, DMODEL, 0.125f, MODE_BF16);
  gemm_bf16_kernel<<<dim3(DMODEL / 128, N_NODES / 128), 256, 0, stream>>>(
      xb, WkT, (const float*)nullptr, Kb, N_NODES, DMODEL, DMODEL, 1.0f, MODE_BF16);
  gemm_bf16_kernel<<<dim3(DMODEL / 128, N_NODES / 128), 256, 0, stream>>>(
      xb, WvT, (const float*)nullptr, Vb, N_NODES, DMODEL, DMODEL, 1.0f, MODE_BF16);

  // biased attention
  attn_kernel<<<NHEAD * (N_NODES / 128), 256, 0, stream>>>(Qb, Kb, Vb, eidx, ebias, Ob);

  // output projection + bias
  gemm_bf16_kernel<<<dim3(DMODEL / 128, N_NODES / 128), 256, 0, stream>>>(
      Ob, WoT, bo, proj, N_NODES, DMODEL, DMODEL, 1.0f, MODE_F32);

  // h = LN(x + proj)
  add_ln_kernel<<<N_NODES, 256, 0, stream>>>(x, proj, g1, be1, hf, hb);

  // FFN
  gemm_bf16_kernel<<<dim3(DFF / 128, N_NODES / 128), 256, 0, stream>>>(
      hb, w1T, b1, ff1, N_NODES, DFF, DMODEL, 1.0f, MODE_BF16_GELU);
  gemm_bf16_kernel<<<dim3(DMODEL / 128, N_NODES / 128), 256, 0, stream>>>(
      ff1, w2T, b2, ff2, N_NODES, DMODEL, DFF, 1.0f, MODE_F32);

  // out = LN(h + ff2)
  add_ln_kernel<<<N_NODES, 256, 0, stream>>>(hf, ff2, g2, be2, (float*)d_out, (u16*)nullptr);

  (void)in_sizes; (void)n_in; (void)out_size; (void)ws_size;
}
